// BipartiteGraphAutoEncoder_ReportBased_45586782879829
// MI455X (gfx1250) — matrix operand order repaired and checked by reading, not compiled, not run on hardware
//
#include <hip/hip_runtime.h>
#include <hip/hip_bf16.h>

// ---------------- CDNA5 WMMA types/helpers ----------------
typedef __attribute__((ext_vector_type(16))) __bf16 v16bf;
typedef __attribute__((ext_vector_type(8)))  float  v8f;

// native f32 -> bf16 conversion (hardware v_cvt on gfx1250, RNE)
__device__ __forceinline__ __bf16 f2bf(float f) { return (__bf16)f; }

__device__ __forceinline__ v16bf zero_frag() {
  v16bf t;
#pragma unroll
  for (int e = 0; e < 16; ++e) t[e] = (__bf16)0.0f;
  return t;
}

// A-matrix 16x32 bf16 fragment from a row-major f32 row pointer.
// lane L: m = L%16 (row chosen by caller), elements e<8 -> K = kbase + (L/16)*8 + e
//         elements e>=8 -> K = kbase + 16 + (L/16)*8 + (e-8)
__device__ __forceinline__ v16bf load_a_f32(const float* __restrict__ row, int kbase, int lane) {
  const float* p = row + kbase + ((lane >> 4) << 3);
  v16bf t;
#pragma unroll
  for (int e = 0; e < 8; ++e) t[e] = f2bf(p[e]);
#pragma unroll
  for (int e = 0; e < 8; ++e) t[e + 8] = f2bf(p[e + 16]);
  return t;
}

// Same A layout but from bf16 row-major LDS row.
__device__ __forceinline__ v16bf load_a_lds(const __bf16* __restrict__ row, int kbase, int lane) {
  const __bf16* p = row + kbase + ((lane >> 4) << 3);
  v16bf t;
#pragma unroll
  for (int e = 0; e < 8; ++e) t[e] = p[e];
#pragma unroll
  for (int e = 0; e < 8; ++e) t[e + 8] = p[e + 16];
  return t;
}

// B-matrix 32x16 fragment for Y = X @ W^T: B[k,n] = W[obase+n, k], W row-major [HOUT][HIN] in LDS (bf16).
// lane L: n = L%16, elements e<8: K = kbase + (L/16)*8 + e; e>=8: K = kbase + 16 + (L/16)*8 + (e-8)
__device__ __forceinline__ v16bf load_b_lds(const __bf16* __restrict__ W, int hin,
                                            int obase, int kbase, int lane) {
  const __bf16* p = W + (size_t)(obase + (lane & 15)) * hin + kbase + ((lane >> 4) << 3);
  v16bf t;
#pragma unroll
  for (int e = 0; e < 8; ++e) t[e] = p[e];
#pragma unroll
  for (int e = 0; e < 8; ++e) t[e + 8] = p[e + 16];
  return t;
}

__device__ __forceinline__ float eluf(float x) { return x > 0.f ? x : (expf(x) - 1.f); }

__device__ __forceinline__ void atomicMaxFloat(float* addr, float val) {
  if (__float_as_int(val) >= 0) {
    atomicMax((int*)addr, __float_as_int(val));
  } else {
    atomicMin((unsigned int*)addr, __float_as_uint(val));
  }
}

// ---------------- Generic fused dense layer: Y[N,HOUT] = act(X[N,HIN] @ W^T + b) ----------------
// 128 threads = 4 waves; each wave owns a 16-row tile; W staged in LDS as bf16.
template <int HIN, int HOUT, int ACT>  // ACT: 0=none 1=relu 2=elu
__global__ __launch_bounds__(128) void k_dense(const float* __restrict__ X,
                                               const float* __restrict__ W,
                                               const float* __restrict__ bias,
                                               float* __restrict__ Y, int N) {
  __shared__ __bf16 Wl[HOUT * HIN];
  __shared__ float bl[HOUT];
  const int tid = threadIdx.x;
  for (int i = tid; i < HOUT * HIN; i += 128) Wl[i] = f2bf(W[i]);
  for (int i = tid; i < HOUT; i += 128) bl[i] = bias ? bias[i] : 0.f;
  __syncthreads();

  const int lane = tid & 31, wave = tid >> 5;
  const int mbase = (blockIdx.x * 4 + wave) * 16;
  const int mrow = mbase + (lane & 15);
  constexpr int KS = HIN / 32;

  v16bf a[KS];
  if (mrow < N) {
    const float* row = X + (size_t)mrow * HIN;
#pragma unroll
    for (int s = 0; s < KS; ++s) a[s] = load_a_f32(row, s * 32, lane);
  } else {
#pragma unroll
    for (int s = 0; s < KS; ++s) a[s] = zero_frag();
  }

#pragma unroll
  for (int ot = 0; ot < HOUT / 16; ++ot) {
    v8f c;
#pragma unroll
    for (int i = 0; i < 8; ++i) c[i] = 0.f;
#pragma unroll
    for (int s = 0; s < KS; ++s) {
      v16bf b = load_b_lds(Wl, HIN, ot * 16, s * 32, lane);
      c = __builtin_amdgcn_wmma_f32_16x16x32_bf16(false, a[s], false, b, (short)0, c, false, false);
    }
    const int n = lane & 15;
    const int mo = mbase + ((lane >> 4) << 3);
    const float bv = bl[ot * 16 + n];
#pragma unroll
    for (int r = 0; r < 8; ++r) {
      const int m = mo + r;
      if (m < N) {
        float v = c[r] + bv;
        if (ACT == 1) v = fmaxf(v, 0.f);
        if (ACT == 2) v = eluf(v);
        Y[(size_t)m * HOUT + ot * 16 + n] = v;
      }
    }
  }
}

// ---------------- per-node attention scalar: s[i] = dot(X[i,:128], a) ----------------
__global__ __launch_bounds__(256) void k_scores(const float* __restrict__ X,
                                                const float* __restrict__ a,
                                                float* __restrict__ s, int N) {
  const int lane = threadIdx.x & 31, wave = threadIdx.x >> 5;
  const int node = blockIdx.x * 8 + wave;
  if (node >= N) return;
  const float* row = X + (size_t)node * 128;
  float acc = 0.f;
#pragma unroll
  for (int j = 0; j < 4; ++j) acc += row[lane + 32 * j] * a[lane + 32 * j];
#pragma unroll
  for (int off = 16; off; off >>= 1) acc += __shfl_xor(acc, off, 32);
  if (lane == 0) s[node] = acc;
}

// v[k] = sum_h aedge[h] * We[h,k]  (We: [128][8])
__global__ void k_vec8(const float* __restrict__ We, const float* __restrict__ aedge,
                       float* __restrict__ v) {
  const int t = threadIdx.x;
  if (t < 8) {
    float s = 0.f;
    for (int h = 0; h < 128; ++h) s += aedge[h] * We[h * 8 + t];
    v[t] = s;
  }
}

__global__ void k_fill(float* __restrict__ p, float val, size_t n) {
  size_t i = (size_t)blockIdx.x * blockDim.x + threadIdx.x;
  size_t stride = (size_t)gridDim.x * blockDim.x;
  for (; i < n; i += stride) p[i] = val;
}

// logits + running segment max
__global__ __launch_bounds__(256) void k_edge_logits(const int* __restrict__ src, const int* __restrict__ dst,
                                                     const float* __restrict__ ea,
                                                     const float* __restrict__ ssrc, const float* __restrict__ sdst,
                                                     const float* __restrict__ v,
                                                     float* __restrict__ alpha, float* __restrict__ amax, int E) {
  const int e = blockIdx.x * 256 + threadIdx.x;
  if (e >= E) return;
  const float* a8 = ea + (size_t)e * 8;
  float dot = 0.f;
#pragma unroll
  for (int k = 0; k < 8; ++k) dot += a8[k] * v[k];
  float x = ssrc[src[e]] + sdst[dst[e]] + dot;
  x = x > 0.f ? x : 0.01f * x;  // leaky_relu(0.01)
  alpha[e] = x;
  atomicMaxFloat(&amax[dst[e]], x);
}

// exp(a - amax[dst]) and segment denominator
__global__ __launch_bounds__(256) void k_edge_expsum(const int* __restrict__ dst,
                                                     float* __restrict__ alpha,
                                                     const float* __restrict__ amax,
                                                     float* __restrict__ den, int E) {
  const int e = blockIdx.x * 256 + threadIdx.x;
  if (e >= E) return;
  const int d = dst[e];
  const float w = expf(alpha[e] - amax[d]);
  alpha[e] = w;
  atomicAdd(&den[d], w);
}

// out[dst] += (alpha/den[dst]) * (xs[src] + edge_attr @ We^T) ; one wave per edge
__global__ __launch_bounds__(256) void k_edge_scatter(const int* __restrict__ src, const int* __restrict__ dst,
                                                      const float* __restrict__ ea, const float* __restrict__ We,
                                                      const float* __restrict__ alpha, const float* __restrict__ den,
                                                      const float* __restrict__ xs, float* __restrict__ out, int E) {
  __shared__ float Wt[8 * 128];  // transposed We: Wt[k][h]
  for (int i = threadIdx.x; i < 1024; i += 256) {
    const int h = i >> 3, k = i & 7;
    Wt[k * 128 + h] = We[i];
  }
  __syncthreads();
  const int lane = threadIdx.x & 31, wave = threadIdx.x >> 5;
  const int e = blockIdx.x * 8 + wave;
  if (e >= E) return;
  const int s = src[e], d = dst[e];
  const float w = alpha[e] / den[d];
  float a8[8];
  const float* pe = ea + (size_t)e * 8;
#pragma unroll
  for (int k = 0; k < 8; ++k) a8[k] = pe[k];
  const float* xrow = xs + (size_t)s * 128;
  float* orow = out + (size_t)d * 128;
#pragma unroll
  for (int j = 0; j < 4; ++j) {
    const int h = lane + 32 * j;
    float ep = 0.f;
#pragma unroll
    for (int k = 0; k < 8; ++k) ep += a8[k] * Wt[k * 128 + h];
    atomicAdd(&orow[h], w * (xrow[h] + ep));
  }
}

// h_new = elu(out + h_old)
__global__ void k_finalize(const float* __restrict__ out, const float* __restrict__ hold,
                           float* __restrict__ hnew, size_t n) {
  size_t i = (size_t)blockIdx.x * blockDim.x + threadIdx.x;
  size_t stride = (size_t)gridDim.x * blockDim.x;
  for (; i < n; i += stride) hnew[i] = eluf(out[i] + hold[i]);
}

// ---------------- fused edge-attribute decoder ----------------
// per 16-edge tile: z = [z_p[prov] ; z_m[mem]] (128) -> relu(z@W1^T+b1) (128) -> @W2^T+b2 (8)
// 128 threads = 4 waves; hidden staged per-wave in LDS (bf16) to re-read in A-layout.
__global__ __launch_bounds__(128) void k_edge_dec(const float* __restrict__ zp, const float* __restrict__ zm,
                                                  const int* __restrict__ prov, const int* __restrict__ mem,
                                                  const float* __restrict__ W1, const float* __restrict__ b1,
                                                  const float* __restrict__ W2, const float* __restrict__ b2,
                                                  float* __restrict__ outp, int E) {
  __shared__ __bf16 W1l[128 * 128];  // 32 KB
  __shared__ __bf16 W2l[16 * 128];   // 4 KB (rows 8..15 zero-padded)
  __shared__ float b1l[128];
  __shared__ __bf16 hs[4][16 * 128];  // 16 KB
  const int tid = threadIdx.x;
  for (int i = tid; i < 128 * 128; i += 128) W1l[i] = f2bf(W1[i]);
  for (int i = tid; i < 16 * 128; i += 128) W2l[i] = (i < 8 * 128) ? f2bf(W2[i]) : (__bf16)0.0f;
  for (int i = tid; i < 128; i += 128) b1l[i] = b1[i];
  __syncthreads();

  const int lane = tid & 31, wave = tid >> 5;
  const long tile = (long)blockIdx.x * 4 + wave;
  const long e0 = tile * 16;
  if (e0 >= E) return;

  // gather A fragments: K 0..63 from z_p[prov], 64..127 from z_m[mem]
  long er = e0 + (lane & 15);
  if (er >= E) er = E - 1;
  const float* zprow = zp + (size_t)prov[er] * 64;
  const float* zmrow = zm + (size_t)mem[er] * 64;
  v16bf a[4];
  a[0] = load_a_f32(zprow, 0, lane);
  a[1] = load_a_f32(zprow, 32, lane);
  a[2] = load_a_f32(zmrow, 0, lane);
  a[3] = load_a_f32(zmrow, 32, lane);

  __bf16* h = &hs[wave][0];
  const int n = lane & 15;
  const int mo = (lane >> 4) << 3;

  // GEMM1 (128->128) + bias + relu, write hidden to LDS as bf16
#pragma unroll
  for (int ot = 0; ot < 8; ++ot) {
    v8f c;
#pragma unroll
    for (int i = 0; i < 8; ++i) c[i] = 0.f;
#pragma unroll
    for (int s = 0; s < 4; ++s) {
      v16bf b = load_b_lds(W1l, 128, ot * 16, s * 32, lane);
      c = __builtin_amdgcn_wmma_f32_16x16x32_bf16(false, a[s], false, b, (short)0, c, false, false);
    }
    const float bv = b1l[ot * 16 + n];
#pragma unroll
    for (int r = 0; r < 8; ++r)
      h[(size_t)(mo + r) * 128 + ot * 16 + n] = f2bf(fmaxf(c[r] + bv, 0.f));
  }

  // GEMM2 (128->8, padded to 16 cols)
  v16bf a2[4];
  {
    const __bf16* hrow = h + (size_t)(lane & 15) * 128;
#pragma unroll
    for (int s = 0; s < 4; ++s) a2[s] = load_a_lds(hrow, s * 32, lane);
  }
  v8f c;
#pragma unroll
  for (int i = 0; i < 8; ++i) c[i] = 0.f;
#pragma unroll
  for (int s = 0; s < 4; ++s) {
    v16bf b = load_b_lds(W2l, 128, 0, s * 32, lane);
    c = __builtin_amdgcn_wmma_f32_16x16x32_bf16(false, a2[s], false, b, (short)0, c, false, false);
  }
  if (n < 8) {
    const float bv = b2[n];
#pragma unroll
    for (int r = 0; r < 8; ++r) {
      const long ee = e0 + mo + r;
      if (ee < E) outp[ee * 8 + n] = c[r] + bv;
    }
  }
}

// ---------------- host orchestration ----------------
extern "C" void kernel_launch(void* const* d_in, const int* in_sizes, int n_in,
                              void* d_out, int out_size, void* d_ws, size_t ws_size,
                              hipStream_t stream) {
  const int NM = in_sizes[0] / 128;  // DM = 128
  const int NP = in_sizes[1] / 64;   // DP = 64
  const int E = in_sizes[2];

  const float* x_member = (const float*)d_in[0];
  const float* x_provider = (const float*)d_in[1];
  const int* edge_prov = (const int*)d_in[2];
  const int* edge_mem = (const int*)d_in[3];
  const float* edge_attr = (const float*)d_in[4];
  const float* proj_m_W = (const float*)d_in[5];
  const float* proj_m_b = (const float*)d_in[6];
  const float* proj_p_W = (const float*)d_in[7];
  const float* proj_p_b = (const float*)d_in[8];
  const float* conv_Wsrc = (const float*)d_in[9];
  const float* conv_Wdst = (const float*)d_in[10];
  const float* conv_We = (const float*)d_in[11];
  const float* conv_asrc = (const float*)d_in[12];
  const float* conv_adst = (const float*)d_in[13];
  const float* conv_aedge = (const float*)d_in[14];
  const float* final_m_W = (const float*)d_in[15];
  const float* final_m_b = (const float*)d_in[16];
  const float* final_p_W = (const float*)d_in[17];
  const float* final_p_b = (const float*)d_in[18];
  const float* dec_m_W1 = (const float*)d_in[19];
  const float* dec_m_b1 = (const float*)d_in[20];
  const float* dec_m_W2 = (const float*)d_in[21];
  const float* dec_m_b2 = (const float*)d_in[22];
  const float* dec_p_W1 = (const float*)d_in[23];
  const float* dec_p_b1 = (const float*)d_in[24];
  const float* dec_p_W2 = (const float*)d_in[25];
  const float* dec_p_b2 = (const float*)d_in[26];
  const float* dec_e_W1 = (const float*)d_in[27];
  const float* dec_e_b1 = (const float*)d_in[28];
  const float* dec_e_W2 = (const float*)d_in[29];
  const float* dec_e_b2 = (const float*)d_in[30];

  // output regions (tuple flattened in return order)
  float* out = (float*)d_out;
  float* xhat_m = out;                        // NM*128
  float* xhat_p = xhat_m + (size_t)NM * 128;  // NP*64
  float* z_m = xhat_p + (size_t)NP * 64;      // NM*64
  float* z_p = z_m + (size_t)NM * 64;         // NP*64
  float* ehat = z_p + (size_t)NP * 64;        // E*8

  // workspace layout (floats)
  float* ws = (float*)d_ws;
  size_t o = 0;
  float* hm0 = ws + o;    o += (size_t)NM * 128;
  float* hm1 = ws + o;    o += (size_t)NM * 128;
  float* hp0 = ws + o;    o += (size_t)NP * 128;
  float* hp1 = ws + o;    o += (size_t)NP * 128;
  float* bufNM = ws + o;  o += (size_t)NM * 128;  // xd(p->m) / xs(m->p) / decoder hidden (m)
  float* bufNP = ws + o;  o += (size_t)NP * 128;  // xs(p->m) / xd(m->p) / decoder hidden (p)
  float* out_m = ws + o;  o += (size_t)NM * 128;
  float* out_p = ws + o;  o += (size_t)NP * 128;
  float* alpha = ws + o;  o += (size_t)E;
  float* amax = ws + o;   o += (size_t)NM;
  float* den = ws + o;    o += (size_t)NM;
  float* ssrc = ws + o;   o += (size_t)NM;
  float* sdst = ws + o;   o += (size_t)NM;
  float* v8 = ws + o;     o += 8;

  const float NEG_INF = -__builtin_huge_valf();
  auto gDense = [](int N) { return dim3((unsigned)((N + 63) / 64)); };
  const dim3 B128(128), B256(256);
  const dim3 gE256((unsigned)((E + 255) / 256));
  const dim3 gE8((unsigned)((E + 7) / 8));

  // ---- input projections (ELU) ----
  k_dense<128, 128, 2><<<gDense(NM), B128, 0, stream>>>(x_member, proj_m_W, proj_m_b, hm0, NM);
  k_dense<64, 128, 2><<<gDense(NP), B128, 0, stream>>>(x_provider, proj_p_W, proj_p_b, hp0, NP);

  float* hm_cur = hm0; float* hm_nxt = hm1;
  float* hp_cur = hp0; float* hp_nxt = hp1;

  for (int l = 0; l < 2; ++l) {
    for (int d = 0; d < 2; ++d) {
      const int idx = 2 * l + d;
      const float* hsrc = (d == 0) ? hp_cur : hm_cur;
      const float* hdst = (d == 0) ? hm_cur : hp_cur;
      const int Ns = (d == 0) ? NP : NM;
      const int Nd = (d == 0) ? NM : NP;
      const int* sidx = (d == 0) ? edge_prov : edge_mem;
      const int* didx = (d == 0) ? edge_mem : edge_prov;
      float* xs = (d == 0) ? bufNP : bufNM;
      float* xd = (d == 0) ? bufNM : bufNP;
      float* ob = (d == 0) ? out_m : out_p;

      const float* Ws = conv_Wsrc + (size_t)idx * 128 * 128;
      const float* Wd = conv_Wdst + (size_t)idx * 128 * 128;
      const float* We = conv_We + (size_t)idx * 128 * 8;

      k_dense<128, 128, 0><<<gDense(Ns), B128, 0, stream>>>(hsrc, Ws, nullptr, xs, Ns);
      k_dense<128, 128, 0><<<gDense(Nd), B128, 0, stream>>>(hdst, Wd, nullptr, xd, Nd);
      k_scores<<<dim3((unsigned)((Ns + 7) / 8)), B256, 0, stream>>>(xs, conv_asrc + idx * 128, ssrc, Ns);
      k_scores<<<dim3((unsigned)((Nd + 7) / 8)), B256, 0, stream>>>(xd, conv_adst + idx * 128, sdst, Nd);
      k_vec8<<<dim3(1), dim3(32), 0, stream>>>(We, conv_aedge + idx * 128, v8);
      k_fill<<<dim3(256), B256, 0, stream>>>(amax, NEG_INF, (size_t)Nd);
      k_fill<<<dim3(256), B256, 0, stream>>>(den, 0.f, (size_t)Nd);
      k_fill<<<dim3(2048), B256, 0, stream>>>(ob, 0.f, (size_t)Nd * 128);
      k_edge_logits<<<gE256, B256, 0, stream>>>(sidx, didx, edge_attr, ssrc, sdst, v8, alpha, amax, E);
      k_edge_expsum<<<gE256, B256, 0, stream>>>(didx, alpha, amax, den, E);
      k_edge_scatter<<<gE8, B256, 0, stream>>>(sidx, didx, edge_attr, We, alpha, den, xs, ob, E);
    }
    // both convs of this layer consumed the old h; now update with residual + ELU
    k_finalize<<<dim3(2048), B256, 0, stream>>>(out_m, hm_cur, hm_nxt, (size_t)NM * 128);
    k_finalize<<<dim3(512), B256, 0, stream>>>(out_p, hp_cur, hp_nxt, (size_t)NP * 128);
    float* t;
    t = hm_cur; hm_cur = hm_nxt; hm_nxt = t;
    t = hp_cur; hp_cur = hp_nxt; hp_nxt = t;
  }

  // ---- final linears -> z (written straight into d_out) ----
  k_dense<128, 64, 0><<<gDense(NM), B128, 0, stream>>>(hm_cur, final_m_W, final_m_b, z_m, NM);
  k_dense<128, 64, 0><<<gDense(NP), B128, 0, stream>>>(hp_cur, final_p_W, final_p_b, z_p, NP);

  // ---- node attribute decoders (2-layer MLP) ----
  k_dense<64, 128, 1><<<gDense(NM), B128, 0, stream>>>(z_m, dec_m_W1, dec_m_b1, bufNM, NM);
  k_dense<128, 128, 0><<<gDense(NM), B128, 0, stream>>>(bufNM, dec_m_W2, dec_m_b2, xhat_m, NM);
  k_dense<64, 128, 1><<<gDense(NP), B128, 0, stream>>>(z_p, dec_p_W1, dec_p_b1, bufNP, NP);
  k_dense<128, 64, 0><<<gDense(NP), B128, 0, stream>>>(bufNP, dec_p_W2, dec_p_b2, xhat_p, NP);

  // ---- fused edge attribute decoder ----
  const long tiles = ((long)E + 15) / 16;
  k_edge_dec<<<dim3((unsigned)((tiles + 3) / 4)), B128, 0, stream>>>(
      z_p, z_m, edge_prov, edge_mem, dec_e_W1, dec_e_b1, dec_e_W2, dec_e_b2, ehat, E);
}
